// GSOP3d_layer_32401233281199
// MI455X (gfx1250) — matrix operand order repaired and checked by reading, not compiled, not run on hardware
//
#include <hip/hip_runtime.h>
#include <hip/hip_bf16.h>

typedef __attribute__((ext_vector_type(16))) _Float16 v16h;
typedef __attribute__((ext_vector_type(8)))  float    v8f;
typedef int v4i __attribute__((vector_size(16)));

#define B_    8
#define C_    64
#define DR_   64
#define M_    110592           // 48*48*48
#define NST_  864              // M_/128 super-tiles per batch
#define WGPB_ 108              // workgroups per batch (864/108 = 8 supertiles each)
#define GROWS_ 80              // 64 Y rows + 16-row tile holding the ones row
#define XSTRIDE_ 20            // raw X row stride in floats (80B: 16B aligned, bank-spread)

#if __has_builtin(__builtin_amdgcn_global_load_async_to_lds_b128) && \
    __has_builtin(__builtin_amdgcn_s_wait_asynccnt)
#define USE_ASYNC 1
#else
#define USE_ASYNC 0
#endif

__device__ __forceinline__ v8f wmma_f16(v16h a, v16h b, v8f c) {
  // 8 args: (neg_a, A, neg_b, B, c_mod, C, reuse_a, reuse_b)
  return __builtin_amdgcn_wmma_f32_16x16x32_f16(false, a, false, b, (short)0, c, false, false);
}

// Issue one wave's 64x16 X slice (raw f32) into its LDS buffer.
// 8 x async B128: lane l -> row c = it*8 + (l>>2), 4-float segment seg = l&3.
__device__ __forceinline__ void issue_xtile(const float* __restrict__ xb, int m0,
                                            float* rawbase, int lane) {
#pragma unroll
  for (int it = 0; it < 8; ++it) {
    int idx = it * 32 + lane;
    int c   = idx >> 2;
    int seg = idx & 3;
    const float* gp = xb + (size_t)c * M_ + (m0 + seg * 4);
    float*       lp = rawbase + c * XSTRIDE_ + seg * 4;
#if USE_ASYNC
    __builtin_amdgcn_global_load_async_to_lds_b128(
        (__attribute__((address_space(1))) v4i*)(uintptr_t)gp,
        (__attribute__((address_space(3))) v4i*)(uintptr_t)lp, 0, 0);
#else
    *(float4*)lp = *(const float4*)gp;
#endif
  }
}

// ---------------------------------------------------------------------------
// Kernel A: fused  Y = relu(W @ relu(X) + b)  and  G += [Y;1] @ Y^T
// grid = B_*WGPB_, block = 256 (8 waves of 32), double-buffered async X stage
// ---------------------------------------------------------------------------
extern "C" __global__ __launch_bounds__(256)
void gsop_fused_gram(const float* __restrict__ x, const float* __restrict__ w_dr,
                     const float* __restrict__ b_dr, float* __restrict__ G) {
  // A-layout f16 fragments of W: [i(4)][kk(2)][lane(32)][h(16)]
  __shared__ __align__(32) _Float16 sWf[4 * 2 * 32 * 16];
  // Raw f32 X slices, per wave, double buffered: [wave(8)][buf(2)][c(64)][XSTRIDE_]
  __shared__ __align__(16) float sXraw[8 * 2 * 64 * XSTRIDE_];
  // Augmented Y (rows 0..79) in A layout: [i(5)][kk(4)][lane][h]
  __shared__ __align__(32) _Float16 sYA[5 * 4 * 32 * 16];
  // Y (rows 0..63) in B layout: [j(4)][kk(4)][lane][h]
  __shared__ __align__(32) _Float16 sYB[4 * 4 * 32 * 16];
  __shared__ float sBias[64];

  const int tid  = threadIdx.x;
  const int lane = tid & 31;
  const int wv   = tid >> 5;
  const int b    = blockIdx.x / WGPB_;
  const int g    = blockIdx.x % WGPB_;

  // ---- preload W (f32 -> f16, A-fragment layout) --------------------------
  for (int e = tid; e < 64 * 64; e += 256) {
    int d  = e >> 6, c = e & 63;
    float v = w_dr[d * 64 + c];
    int i  = d >> 4, rr = d & 15;
    int kk = c >> 5, co = c & 31;
    int ln = rr + (((co >> 3) & 1) << 4);
    int hh = (((co >> 4) & 1) << 3) + (co & 7);
    sWf[(((i * 2 + kk) * 32 + ln) << 4) + hh] = (_Float16)v;
  }
  if (tid < 64) sBias[tid] = b_dr[tid];
  // constant ones-row tile (augmented rows 64..79): row 64 = 1, rows 65..79 = 0
  for (int e = tid; e < 4 * 32 * 16; e += 256) {
    int kk = e >> 9, ln = (e >> 4) & 31, hh = e & 15;
    sYA[(((4 * 4 + kk) * 32 + ln) << 4) + hh] =
        ((ln & 15) == 0) ? (_Float16)1.0f : (_Float16)0.0f;
  }
  __syncthreads();

  const float* xb = x + (size_t)b * C_ * M_;
  v8f acc0 = {}, acc1 = {}, acc2 = {};
  const int t0 = wv, t1 = wv + 8, t2 = wv + 16;

  // prologue: start async fill of buffer 0 for the first super-tile
  float* raw0 = &sXraw[(wv * 2 + 0) * 64 * XSTRIDE_];
  float* raw1 = &sXraw[(wv * 2 + 1) * 64 * XSTRIDE_];
  issue_xtile(xb, g * 128 + wv * 16, raw0, lane);

  int buf = 0;
  for (int st = g; st < NST_; st += WGPB_) {
    const bool hasNext = (st + WGPB_) < NST_;
    // prefetch next super-tile into the other buffer while we compute
    if (hasNext)
      issue_xtile(xb, (st + WGPB_) * 128 + wv * 16, buf ? raw0 : raw1, lane);
#if USE_ASYNC
    if (hasNext) __builtin_amdgcn_s_wait_asynccnt(8);  // drain older batch only
    else         __builtin_amdgcn_s_wait_asynccnt(0);
#endif
    const float* raw = buf ? raw1 : raw0;

    // ---- stage 1: Y tile = W @ relu(X); gather B-frags straight from raw LDS
    v8f y0 = {}, y1 = {}, y2 = {}, y3 = {};
    const int mloc  = lane & 15;
    const int cbase = (lane >> 4) << 3;   // +8 for high half-lanes (B layout)
#pragma unroll
    for (int kk = 0; kk < 2; ++kk) {
      v16h bf;
#pragma unroll
      for (int h = 0; h < 16; ++h) {
        float f = raw[(kk * 32 + cbase + (h & 7) + ((h >> 3) << 4)) * XSTRIDE_ + mloc];
        f = f > 0.f ? f : 0.f;            // relu on X
        bf[h] = (_Float16)f;
      }
      const v16h a0 = *(const v16h*)&sWf[((0 * 2 + kk) * 32 + lane) << 4];
      const v16h a1 = *(const v16h*)&sWf[((1 * 2 + kk) * 32 + lane) << 4];
      const v16h a2 = *(const v16h*)&sWf[((2 * 2 + kk) * 32 + lane) << 4];
      const v16h a3 = *(const v16h*)&sWf[((3 * 2 + kk) * 32 + lane) << 4];
      y0 = wmma_f16(a0, bf, y0);
      y1 = wmma_f16(a1, bf, y1);
      y2 = wmma_f16(a2, bf, y2);
      y3 = wmma_f16(a3, bf, y3);
    }

    // ---- stage 1c: bias + relu, write Y in both A and B fragment layouts
    {
      const int mo  = wv * 16 + (lane & 15);  // column offset 0..127 in supertile
      const int kk2 = mo >> 5;
      const int moo = mo & 31;
      const int lnA = ((moo >> 3) & 1) << 4;
      const int hhA = (((moo >> 4) & 1) << 3) + (moo & 7);
      const int lnB = ((moo >> 3) & 1) << 4;
      const int hhB = (((moo >> 4) & 1) << 3) + (moo & 7);
      const int rhi = (lane >> 4) << 3;
#pragma unroll
      for (int i = 0; i < 4; ++i) {
        v8f y = (i == 0) ? y0 : ((i == 1) ? y1 : ((i == 2) ? y2 : y3));
#pragma unroll
        for (int r = 0; r < 8; ++r) {
          int row  = i * 16 + r + rhi;
          float vv = y[r] + sBias[row];
          vv = vv > 0.f ? vv : 0.f;
          _Float16 hv = (_Float16)vv;
          int rl = row & 15;
          sYA[(((i * 4 + kk2) * 32 + (rl + lnA)) << 4) + hhA] = hv;
          sYB[(((i * 4 + kk2) * 32 + (rl + lnB)) << 4) + hhB] = hv;
        }
      }
    }
    __syncthreads();

    // ---- stage 2: Gram tiles G_ext[i,j] += Ytile_i @ Ytile_j^T (K = 128)
    {
      int i0 = t0 >> 2, j0 = t0 & 3;
      int i1 = t1 >> 2, j1 = t1 & 3;
#pragma unroll
      for (int kk = 0; kk < 4; ++kk) {
        v16h a  = *(const v16h*)&sYA[((i0 * 4 + kk) * 32 + lane) << 4];
        v16h bb = *(const v16h*)&sYB[((j0 * 4 + kk) * 32 + lane) << 4];
        acc0 = wmma_f16(a, bb, acc0);
      }
#pragma unroll
      for (int kk = 0; kk < 4; ++kk) {
        v16h a  = *(const v16h*)&sYA[((i1 * 4 + kk) * 32 + lane) << 4];
        v16h bb = *(const v16h*)&sYB[((j1 * 4 + kk) * 32 + lane) << 4];
        acc1 = wmma_f16(a, bb, acc1);
      }
      if (wv < 4) {
        int i2 = t2 >> 2, j2 = t2 & 3;
#pragma unroll
        for (int kk = 0; kk < 4; ++kk) {
          v16h a  = *(const v16h*)&sYA[((i2 * 4 + kk) * 32 + lane) << 4];
          v16h bb = *(const v16h*)&sYB[((j2 * 4 + kk) * 32 + lane) << 4];
          acc2 = wmma_f16(a, bb, acc2);
        }
      }
    }
    __syncthreads();
    buf ^= 1;
  }

  // ---- flush register-resident partial Gram with global f32 atomics -------
  float* Gb = G + b * GROWS_ * 64;
  const int rhi  = (lane >> 4) << 3;
  const int col0 = lane & 15;
  {
    int i = t0 >> 2, j = t0 & 3;
#pragma unroll
    for (int r = 0; r < 8; ++r)
      atomicAdd(&Gb[(i * 16 + r + rhi) * 64 + j * 16 + col0], acc0[r]);
  }
  {
    int i = t1 >> 2, j = t1 & 3;
#pragma unroll
    for (int r = 0; r < 8; ++r)
      atomicAdd(&Gb[(i * 16 + r + rhi) * 64 + j * 16 + col0], acc1[r]);
  }
  if (wv < 4) {
    int i = t2 >> 2, j = t2 & 3;
#pragma unroll
    for (int r = 0; r < 8; ++r)
      atomicAdd(&Gb[(i * 16 + r + rhi) * 64 + j * 16 + col0], acc2[r]);
  }
}

// ---------------------------------------------------------------------------
// Kernel B: cov -> grouped row conv -> FC -> sigmoid gates (tiny)
// ---------------------------------------------------------------------------
extern "C" __global__ __launch_bounds__(64)
void gsop_reduce(const float* __restrict__ G, const float* __restrict__ w_row,
                 const float* __restrict__ b_row, const float* __restrict__ w_fc,
                 const float* __restrict__ b_fc, float* __restrict__ sgate) {
  __shared__ float rsh[512];
  const int b = blockIdx.x, d = threadIdx.x;
  const float* Gb  = G + b * GROWS_ * 64;
  const float invM = 1.0f / (float)M_;
  const float Sd   = Gb[64 * 64 + d];        // row 64 of augmented Gram = row sums
  float acc[8];
#pragma unroll
  for (int j = 0; j < 8; ++j) acc[j] = b_row[d * 8 + j];
  for (int k = 0; k < 64; ++k) {
    float Sk  = Gb[64 * 64 + k];
    float cov = Gb[d * 64 + k] * invM - Sd * Sk * invM * invM;
#pragma unroll
    for (int j = 0; j < 8; ++j) acc[j] += cov * w_row[d * 512 + j * 64 + k];
  }
#pragma unroll
  for (int j = 0; j < 8; ++j) rsh[d * 8 + j] = acc[j];
  __syncthreads();
  float a = b_fc[d];
  for (int o = 0; o < 512; ++o) a += rsh[o] * w_fc[d * 512 + o];
  sgate[b * 64 + d] = 1.0f / (1.0f + __expf(-a));
}

// ---------------------------------------------------------------------------
// Kernel C: out = x * s[b,c]  (pure bandwidth, float4)
// ---------------------------------------------------------------------------
extern "C" __global__ __launch_bounds__(256)
void gsop_scale(const float4* __restrict__ x4, const float* __restrict__ sgate,
                float4* __restrict__ o4) {
  int n = blockIdx.x * 256 + threadIdx.x;
  const int m4 = M_ / 4;
  if (n < B_ * C_ * m4) {
    float s  = sgate[n / m4];   // index == b*64 + c
    float4 v = x4[n];
    v.x *= s; v.y *= s; v.z *= s; v.w *= s;
    o4[n] = v;
  }
}

extern "C" __global__ void gsop_zero(float* __restrict__ G) {
  int n = blockIdx.x * 256 + threadIdx.x;
  if (n < B_ * GROWS_ * 64) G[n] = 0.0f;
}

// ---------------------------------------------------------------------------
extern "C" void kernel_launch(void* const* d_in, const int* in_sizes, int n_in,
                              void* d_out, int out_size, void* d_ws, size_t ws_size,
                              hipStream_t stream) {
  (void)in_sizes; (void)n_in; (void)out_size; (void)ws_size;
  const float* x     = (const float*)d_in[0];
  const float* w_dr  = (const float*)d_in[1];
  const float* b_dr  = (const float*)d_in[2];
  const float* w_row = (const float*)d_in[3];
  const float* b_row = (const float*)d_in[4];
  const float* w_fc  = (const float*)d_in[5];
  const float* b_fc  = (const float*)d_in[6];

  float* G     = (float*)d_ws;               // B*80*64 f32 = 160 KB
  float* sgate = G + B_ * GROWS_ * 64;       // B*64 f32

  hipLaunchKernelGGL(gsop_zero, dim3((B_ * GROWS_ * 64 + 255) / 256), dim3(256),
                     0, stream, G);
  hipLaunchKernelGGL(gsop_fused_gram, dim3(B_ * WGPB_), dim3(256), 0, stream,
                     x, w_dr, b_dr, G);
  hipLaunchKernelGGL(gsop_reduce, dim3(B_), dim3(64), 0, stream,
                     G, w_row, b_row, w_fc, b_fc, sgate);
  const int total4 = B_ * C_ * (M_ / 4);
  hipLaunchKernelGGL(gsop_scale, dim3((total4 + 255) / 256), dim3(256), 0, stream,
                     (const float4*)x, sgate, (float4*)d_out);
}
